// sLSTMLayer_46033459478751
// MI455X (gfx1250) — matrix operand order repaired and checked by reading, not compiled, not run on hardware
//
#include <hip/hip_runtime.h>
#include <math.h>

// sLSTM layer for MI455X (gfx1250, wave32, WMMA).
// conv+swish -> WMMA gate projections -> per-head sequential scan
// (WMMA recurrent matmul, weights L2-resident as pre-split bf16 fragments)
// -> per-head groupnorm.  bf16 "hi+lo" split (3 WMMAs / K-tile) gives
// ~f32 accuracy at ~2.7x the per-FLOP rate of the native f32 16x16x4 path.
// All bf16 conversion work is hoisted OUT of the WMMA inner loops: operands
// are staged in LDS pre-split, so fragment builds are pure ds_load_b128.

#define B_ 4
#define S_ 2048
#define D_ 1024
#define NH_ 4
#define DH_ 256
#define KCONV 4
#define EPSV 1e-6f

typedef __attribute__((ext_vector_type(16))) __bf16 v16bf;
typedef __attribute__((ext_vector_type(16))) unsigned short v16u;
typedef __attribute__((ext_vector_type(8)))  unsigned short v8us;
typedef __attribute__((ext_vector_type(8)))  float v8f;

union Frag16 { v16u u; v8us h[2]; v16bf b; };

__device__ __forceinline__ unsigned short bf16_rn(float f) {
  union { float f; unsigned u; } c; c.f = f;
  unsigned u = c.u;
  u += 0x7fffu + ((u >> 16) & 1u);      // round-to-nearest-even
  return (unsigned short)(u >> 16);
}
__device__ __forceinline__ float bf16_f(unsigned short h) {
  union { unsigned u; float f; } c; c.u = ((unsigned)h) << 16;
  return c.f;
}

// Build an A fragment (16x32 bf16) from row-major bf16 LDS rows.
// A half-layout: lane<16: j0..7 -> K0..7,  j8..15 -> K16..23
//                lane>=16: j0..7 -> K8..15, j8..15 -> K24..31
// => two contiguous 8-half (16B) runs at k0 and k0+16, k0 = kT*32+8*(lane>=16).
__device__ __forceinline__ Frag16 load_afrag(const unsigned short* row,
                                             int k0) {
  Frag16 a;
  a.h[0] = *(const v8us*)(row + k0);
  a.h[1] = *(const v8us*)(row + k0 + 16);
  return a;
}

// ---------------------------------------------------------------------------
// Weight prep: split f32 weights into bf16 hi/lo B-matrix fragments, stored
// fragment-major so the hot loops do contiguous 32B loads per lane.
// B-layout (K=32, 16-bit): lane holds column n=lane&15; half j holds
// K = (j&7) + 8*(lane>=16) + 16*(j>=8).  Blob = 1024 halves: [hi:512][lo:512], lane*16+j.
// ---------------------------------------------------------------------------
__global__ void prep_proj_frags(const float* __restrict__ wi,
                                const float* __restrict__ wf,
                                const float* __restrict__ wz,
                                const float* __restrict__ wo,
                                unsigned short* __restrict__ frag) {
  int tid = blockIdx.x * blockDim.x + threadIdx.x;   // 65536 threads
  int lane = tid & 31;
  int f = tid >> 5;                                  // 2048 fragments
  int kT = f & 7;
  int nT = (f >> 3) & 15;
  int g  = (f >> 7) & 3;
  int h  = f >> 9;
  const float* W = (g == 0) ? wi : (g == 1) ? wf : (g == 2) ? wz : wo;
  int e = nT * 16 + (lane & 15);
  unsigned short* hi = frag + (size_t)f * 1024 + lane * 16;
  unsigned short* lo = hi + 512;
  for (int j = 0; j < 16; ++j) {
    int k = kT * 32 + (j & 7) + ((lane & 16) >> 1) + ((j & 8) << 1);
    float v = W[((size_t)h * DH_ + k) * DH_ + e];
    unsigned short hb = bf16_rn(v);
    hi[j] = hb;
    lo[j] = bf16_rn(v - bf16_f(hb));
  }
}

__global__ void prep_rec_frags(const float* __restrict__ rk,
                               unsigned short* __restrict__ frag) {
  int tid = blockIdx.x * blockDim.x + threadIdx.x;   // 65536 threads
  int lane = tid & 31;
  int f = tid >> 5;                                  // 2048 fragments
  int kT = f & 7;
  int nT = (f >> 3) & 63;                            // N = 4 gates * 256
  int h  = f >> 9;
  int ng = nT * 16 + (lane & 15);
  int g = ng >> 8, e = ng & 255;
  unsigned short* hi = frag + (size_t)f * 1024 + lane * 16;
  unsigned short* lo = hi + 512;
  for (int j = 0; j < 16; ++j) {
    int k = kT * 32 + (j & 7) + ((lane & 16) >> 1) + ((j & 8) << 1);
    float v = rk[(((size_t)h * DH_ + k) * 4 + g) * DH_ + e];
    unsigned short hb = bf16_rn(v);
    hi[j] = hb;
    lo[j] = bf16_rn(v - bf16_f(hb));
  }
}

// ---------------------------------------------------------------------------
// Causal depthwise conv (K=4) + swish.  Pure streaming, memory-bound.
// ---------------------------------------------------------------------------
__global__ void conv_swish(const float* __restrict__ x,
                           const float* __restrict__ ck,
                           const float* __restrict__ cb,
                           float* __restrict__ xc) {
  int t = blockIdx.x * 256 + threadIdx.x;
  int d = t & (D_ - 1);
  int s = (t >> 10) & (S_ - 1);
  int b = t >> 21;
  float acc = cb[d];
#pragma unroll
  for (int k = 0; k < KCONV; ++k) {
    int ss = s + k - (KCONV - 1);
    if (ss >= 0)
      acc += x[((size_t)(b * S_ + ss)) * D_ + d] * ck[k * D_ + d];
  }
  xc[t] = acc / (1.f + __expf(-acc));
}

// ---------------------------------------------------------------------------
// Gate projections: per (head, gate) GEMM  [8192 x 256] @ [256 x 256].
// Block = (mTile, h*4+g); 16 waves, one 16x16 N-tile each; K=256 -> 8 K-tiles
// x 3 split-bf16 WMMAs = 24 v_wmma per wave.  A tile is pre-split to bf16
// hi/lo in LDS, so fragment builds are pure ds_load_b128.
// ---------------------------------------------------------------------------
__global__ __launch_bounds__(512, 1)
void proj_wmma(const float* __restrict__ x, const float* __restrict__ xc,
               const unsigned short* __restrict__ frag,
               float* __restrict__ gates) {
  int mT = blockIdx.x;                // 512 row tiles of B*S
  int hg = blockIdx.y;                // 16 = NH*4
  int h = hg >> 2, g = hg & 3;
  int tid = threadIdx.x;
  int wave = tid >> 5, lane = tid & 31;
  const float* src = (g < 2) ? xc : x;   // i,f use conv output; z,o use x

  __shared__ unsigned short sAhi[16 * 256];   // 8 KB
  __shared__ unsigned short sAlo[16 * 256];   // 8 KB
  for (int q = 0; q < 8; ++q) {
    int i = q * 512 + tid;
    int row = i >> 8, col = i & 255;
    float v = src[((size_t)(mT * 16 + row)) * D_ + h * DH_ + col];
    unsigned short hb = bf16_rn(v);
    sAhi[i] = hb;
    sAlo[i] = bf16_rn(v - bf16_f(hb));
  }
  __syncthreads();

  int nT = wave;
  int m = lane & 15;
  const unsigned short* rowhi = sAhi + m * 256;
  const unsigned short* rowlo = sAlo + m * 256;
  v8f acc = {};
  for (int kT = 0; kT < 8; ++kT) {
    int k0 = kT * 32 + ((lane & 16) >> 1);
    Frag16 ahi = load_afrag(rowhi, k0);
    Frag16 alo = load_afrag(rowlo, k0);
    size_t fidx = (((size_t)(h * 4 + g) * 16 + nT) * 8 + kT) * 1024;
    Frag16 bhi, blo;
    bhi.u = *(const v16u*)(frag + fidx + lane * 16);
    blo.u = *(const v16u*)(frag + fidx + 512 + lane * 16);
    acc = __builtin_amdgcn_wmma_f32_16x16x32_bf16(false, ahi.b, false, bhi.b,
                                                  (short)0, acc, false, false);
    acc = __builtin_amdgcn_wmma_f32_16x16x32_bf16(false, ahi.b, false, blo.b,
                                                  (short)0, acc, false, false);
    acc = __builtin_amdgcn_wmma_f32_16x16x32_bf16(false, alo.b, false, bhi.b,
                                                  (short)0, acc, false, false);
  }
  // D layout: n = lane&15, m = r + 8*(lane>=16); store gates[s][b][g][h][e]
  int e = nT * 16 + (lane & 15);
#pragma unroll
  for (int r = 0; r < 8; ++r) {
    int mm = r + ((lane & 16) >> 1);
    int gr = mT * 16 + mm;
    int b = gr >> 11, s = gr & (S_ - 1);
    gates[((((size_t)s * B_ + b) * 4 + g) * NH_ + h) * DH_ + e] = acc[r];
  }
}

// ---------------------------------------------------------------------------
// Sequential sLSTM scan: one workgroup per head, B=4 chains batched into a
// zero-padded M=16 WMMA tile.  16 waves x 4 N-tiles cover N=1024 (4 gates).
// Recurrent weight fragments (4 MB total) stay L2-resident.  Y state lives in
// LDS pre-split as bf16 hi/lo (rows 4..15 zeroed once), so the per-step GEMM
// does zero conversion ALU: 4 ds_load_b128 + 8 global 32B loads + 12 WMMAs
// per (wave, kT).
// ---------------------------------------------------------------------------
__global__ __launch_bounds__(512, 1)
void slstm_scan(const float* __restrict__ gates,
                const unsigned short* __restrict__ rfrag,
                const float* __restrict__ rbias,
                float* __restrict__ out) {
  int h = blockIdx.x;
  int tid = threadIdx.x;
  int wave = tid >> 5, lane = tid & 31;

  __shared__ unsigned short sYhi[16 * 256];   // padded A tile, hi halves
  __shared__ unsigned short sYlo[16 * 256];   // padded A tile, lo halves
  __shared__ float sRaw[4 * 1024];            // raw gate pre-activations

  for (int i = tid; i < 16 * 256; i += 512) { sYhi[i] = 0; sYlo[i] = 0; }

  // Two cell slots per thread: (b0,e) and (b0+2,e).
  int b0 = tid >> 8, e = tid & 255;
  int b1 = b0 + 2;
  float c0 = 0.f, n0 = 0.f, m0 = 0.f;
  float c1 = 0.f, n1 = 0.f, m1 = 0.f;
  float bI = rbias[(0 * NH_ + h) * DH_ + e];
  float bF = rbias[(1 * NH_ + h) * DH_ + e];
  float bZ = rbias[(2 * NH_ + h) * DH_ + e];
  float bO = rbias[(3 * NH_ + h) * DH_ + e];

  int m = lane & 15;
  const unsigned short* rowhi = sYhi + m * 256;
  const unsigned short* rowlo = sYlo + m * 256;
  __syncthreads();

  for (int s = 0; s < S_; ++s) {
    // Issue this step's gate loads early; overlap with WMMA below.
    const float* gb = gates + (size_t)s * B_ * 4 * NH_ * DH_;
    float gi0 = gb[((b0 * 4 + 0) * NH_ + h) * DH_ + e];
    float gf0 = gb[((b0 * 4 + 1) * NH_ + h) * DH_ + e];
    float gz0 = gb[((b0 * 4 + 2) * NH_ + h) * DH_ + e];
    float go0 = gb[((b0 * 4 + 3) * NH_ + h) * DH_ + e];
    float gi1 = gb[((b1 * 4 + 0) * NH_ + h) * DH_ + e];
    float gf1 = gb[((b1 * 4 + 1) * NH_ + h) * DH_ + e];
    float gz1 = gb[((b1 * 4 + 2) * NH_ + h) * DH_ + e];
    float go1 = gb[((b1 * 4 + 3) * NH_ + h) * DH_ + e];
    if (s + 1 < S_)                    // global_prefetch_b8 for next step
      __builtin_prefetch(gb + (size_t)B_ * 4 * NH_ * DH_, 0, 0);

    // raw = Ypad(16x256) @ Wrec(256x1024)
    v8f acc[4] = {};
    for (int kT = 0; kT < 8; ++kT) {
      int k0 = kT * 32 + ((lane & 16) >> 1);
      Frag16 ahi = load_afrag(rowhi, k0);
      Frag16 alo = load_afrag(rowlo, k0);
#pragma unroll
      for (int q = 0; q < 4; ++q) {
        int nT = wave * 4 + q;
        size_t fidx = (((size_t)h * 64 + nT) * 8 + kT) * 1024;
        Frag16 bhi, blo;
        bhi.u = *(const v16u*)(rfrag + fidx + lane * 16);
        blo.u = *(const v16u*)(rfrag + fidx + 512 + lane * 16);
        acc[q] = __builtin_amdgcn_wmma_f32_16x16x32_bf16(
            false, ahi.b, false, bhi.b, (short)0, acc[q], false, false);
        acc[q] = __builtin_amdgcn_wmma_f32_16x16x32_bf16(
            false, ahi.b, false, blo.b, (short)0, acc[q], false, false);
        acc[q] = __builtin_amdgcn_wmma_f32_16x16x32_bf16(
            false, alo.b, false, bhi.b, (short)0, acc[q], false, false);
      }
    }
    if (lane < 16) {                   // only m=0..3 rows are real batches
      int n = lane & 15;
#pragma unroll
      for (int q = 0; q < 4; ++q) {
        int nT = wave * 4 + q;
#pragma unroll
        for (int r = 0; r < 4; ++r)
          sRaw[r * 1024 + nT * 16 + n] = acc[q][r];
      }
    }
    __syncthreads();

    // sLSTM cell update (exponential gating with max-stabilizer).
#pragma unroll
    for (int slot = 0; slot < 2; ++slot) {
      int b = slot ? b1 : b0;
      float ir = sRaw[b * 1024 + 0 * 256 + e] + (slot ? gi1 : gi0) + bI;
      float fr = sRaw[b * 1024 + 1 * 256 + e] + (slot ? gf1 : gf0) + bF;
      float zr = sRaw[b * 1024 + 2 * 256 + e] + (slot ? gz1 : gz0) + bZ;
      float orw = sRaw[b * 1024 + 3 * 256 + e] + (slot ? go1 : go0) + bO;
      float& c = slot ? c1 : c0;
      float& n = slot ? n1 : n0;
      float& mm = slot ? m1 : m0;
      float ls = (fr >= 0.f) ? -log1pf(__expf(-fr))
                             : (fr - log1pf(__expf(fr)));
      float lf = mm + ls;
      float mn = fmaxf(ir, lf);
      float ig = __expf(ir - mn);
      float fg = __expf(lf - mn);
      c = fg * c + ig * tanhf(zr);
      n = fg * n + ig;
      mm = mn;
      float y = (c / n) / (1.f + __expf(-orw));
      unsigned short hb = bf16_rn(y);              // pre-split for next step
      sYhi[b * 256 + e] = hb;
      sYlo[b * 256 + e] = bf16_rn(y - bf16_f(hb));
      out[((size_t)(b * S_ + s)) * D_ + h * DH_ + e] = y;
    }
    __syncthreads();
  }
}

// ---------------------------------------------------------------------------
// GroupNorm over each head's DH=256, in place on d_out, then affine.
// wave32 shuffle reduction + one small LDS pass.
// ---------------------------------------------------------------------------
__global__ void groupnorm(float* __restrict__ out,
                          const float* __restrict__ gs,
                          const float* __restrict__ gb) {
  int blk = blockIdx.x;                // B*S*NH
  int h = blk & 3;
  int s = (blk >> 2) & (S_ - 1);
  int b = blk >> 13;
  int e = threadIdx.x;
  int wave = e >> 5, lane = e & 31;
  size_t base = ((size_t)(b * S_ + s)) * D_ + h * DH_;
  float v = out[base + e];
  float sum = v, sq = v * v;
#pragma unroll
  for (int o = 16; o > 0; o >>= 1) {
    sum += __shfl_xor(sum, o, 32);
    sq  += __shfl_xor(sq, o, 32);
  }
  __shared__ float p1[8], p2[8], mv[2];
  if (lane == 0) { p1[wave] = sum; p2[wave] = sq; }
  __syncthreads();
  if (e == 0) {
    float t1 = 0.f, t2 = 0.f;
#pragma unroll
    for (int w = 0; w < 8; ++w) { t1 += p1[w]; t2 += p2[w]; }
    float mu = t1 * (1.f / 256.f);
    mv[0] = mu;
    mv[1] = rsqrtf(t2 * (1.f / 256.f) - mu * mu + EPSV);
  }
  __syncthreads();
  out[base + e] = (v - mv[0]) * mv[1] * gs[h * DH_ + e] + gb[h * DH_ + e];
}

// ---------------------------------------------------------------------------
extern "C" void kernel_launch(void* const* d_in, const int* in_sizes, int n_in,
                              void* d_out, int out_size, void* d_ws,
                              size_t ws_size, hipStream_t stream) {
  const float* x  = (const float*)d_in[0];
  const float* ck = (const float*)d_in[1];
  const float* cb = (const float*)d_in[2];
  const float* wi = (const float*)d_in[3];
  const float* wf = (const float*)d_in[4];
  const float* wz = (const float*)d_in[5];
  const float* wo = (const float*)d_in[6];
  const float* rk = (const float*)d_in[7];
  const float* rb = (const float*)d_in[8];
  const float* gs = (const float*)d_in[9];
  const float* gb = (const float*)d_in[10];
  float* out = (float*)d_out;

  // Workspace layout (~176 MB):
  char* ws = (char*)d_ws;
  float* xc             = (float*)ws;                          //  32 MB
  float* gates          = (float*)(ws + (size_t)33554432);     // 128 MB
  unsigned short* pfrag = (unsigned short*)(ws + (size_t)167772160); // 4 MB
  unsigned short* rfrag = (unsigned short*)(ws + (size_t)171966464); // 4 MB

  prep_proj_frags<<<256, 256, 0, stream>>>(wi, wf, wz, wo, pfrag);
  prep_rec_frags<<<256, 256, 0, stream>>>(rk, rfrag);
  conv_swish<<<(B_ * S_ * D_) / 256, 256, 0, stream>>>(x, ck, cb, xc);
  dim3 pg(512, 16);
  proj_wmma<<<pg, 512, 0, stream>>>(x, xc, pfrag, gates);
  slstm_scan<<<NH_, 512, 0, stream>>>(gates, rfrag, rb, out);
  groupnorm<<<B_ * S_ * NH_, 256, 0, stream>>>(out, gs, gb);
}
